// PointNet2CornerDetection_19816979104315
// MI455X (gfx1250) — matrix operand order repaired and checked by reading, not compiled, not run on hardware
//
#include <hip/hip_runtime.h>

// ---------------------------------------------------------------------------
// Types for CDNA5 WMMA (wave32, 16x16x32 bf16 -> f32)
// ---------------------------------------------------------------------------
typedef __attribute__((ext_vector_type(16))) __bf16 v16bf;
typedef __attribute__((ext_vector_type(8)))  float  v8f;
typedef __attribute__((ext_vector_type(4)))  float  f4;

static __device__ __forceinline__ __bf16 f2bf_rne(float f) {
  union { float f; unsigned u; } in; in.f = f;
  unsigned u = in.u;
  unsigned r = u + 0x7FFFu + ((u >> 16) & 1u);   // round-to-nearest-even
  union { unsigned short s; __bf16 b; } out;
  out.s = (unsigned short)(r >> 16);
  return out.b;
}

// ---------------------------------------------------------------------------
// Weight pre-pack: W[O,C] fp32 -> Wb[O,Cs] bf16, zero-padded K tail.
// ---------------------------------------------------------------------------
__global__ void pack_w_kernel(const float* __restrict__ W, __bf16* __restrict__ Wb,
                              int O, int C, int Cs)
{
  int t = blockIdx.x * blockDim.x + threadIdx.x;
  if (t >= O * Cs) return;
  int o = t / Cs, c = t % Cs;
  float v = (c < C) ? W[(size_t)o * C + c] : 0.0f;
  Wb[t] = f2bf_rne(v);
}

// ---------------------------------------------------------------------------
// Fused GEMM + BN + ReLU:  Y[m,o] = relu( s_o * sum_c X[m,c]*W[o,c] + bias_o )
// Preconditions (guaranteed by the host code):
//   M % 64 == 0, Cs % 32 == 0 (X rows zero-padded to Cs), O % 32 == 0,
//   X 32B-aligned rows, Wb bf16 pre-packed [O, Cs].
// Block = 128 threads (4 waves), block tile 64(M) x 64(O);
// each wave: 64(M) x 16(O) via 4 accumulators, K stepped by 32.
// No lane-divergent control flow -> EXEC all-ones at every WMMA.
// ---------------------------------------------------------------------------
__global__ void __launch_bounds__(128)
wmma_mlp_kernel(const float* __restrict__ X, const __bf16* __restrict__ Wb,
                const float* __restrict__ Bv, const float* __restrict__ Gm,
                const float* __restrict__ Bt, const float* __restrict__ Mn,
                const float* __restrict__ Vr, float* __restrict__ Y,
                int M, int Cs, int O)
{
  const int lane  = threadIdx.x & 31;
  const int wave  = threadIdx.x >> 5;
  const int h     = lane >> 4;   // lane half
  const int q     = lane & 15;
  const int mBase = blockIdx.x * 64;
  const int o0    = blockIdx.y * 64 + wave * 16;
  if (o0 >= O) return;                       // wave-uniform

  const int oc = o0 + q;                     // < O since O % 32 == 0
  const __bf16* Wrow = Wb + (size_t)oc * Cs;
  const float*  Xrow = X + (size_t)(mBase + q) * Cs;
  const size_t  rstep = (size_t)16 * Cs;     // row-tile stride in floats

  v8f acc[4] = {};

  for (int c0 = 0; c0 < Cs; c0 += 32) {
    // B fragment: 16 contiguous bf16 of column oc, K in [c0+16h, c0+16h+16)
    v16bf bf = *(const v16bf*)(Wrow + c0 + 16 * h);
#pragma unroll
    for (int rt = 0; rt < 4; ++rt) {
      const float* Xr = Xrow + rstep * rt + c0;
      // A fragment: 16 contiguous floats, K in [c0+16h, c0+16h+16) -> b128 loads
      const f4* p0 = (const f4*)(Xr + 16 * h);
      const f4* p1 = (const f4*)(Xr + 16 * h + 8);
      f4 x0 = p0[0], x1 = p0[1];
      f4 x2 = p1[0], x3 = p1[1];
      v16bf af;
#pragma unroll
      for (int t = 0; t < 4; ++t) {
        af[t]      = (__bf16)x0[t];
        af[4 + t]  = (__bf16)x1[t];
        af[8 + t]  = (__bf16)x2[t];
        af[12 + t] = (__bf16)x3[t];
      }
      acc[rt] = __builtin_amdgcn_wmma_f32_16x16x32_bf16(
          false, af, false, bf, (short)0, acc[rt], false, false);
    }
  }

  // epilogue: BN fold + ReLU. C/D layout: row = v + 8h, col = q.
  float inv = rsqrtf(Vr[oc] + 1e-5f);
  float s   = Gm[oc] * inv;
  float bb  = (Bv[oc] - Mn[oc]) * s + Bt[oc];
#pragma unroll
  for (int rt = 0; rt < 4; ++rt) {
#pragma unroll
    for (int v = 0; v < 8; ++v) {
      int m = mBase + rt * 16 + 8 * h + v;
      float yv = acc[rt][v] * s + bb;
      Y[(size_t)m * O + oc] = fmaxf(yv, 0.0f);
    }
  }
}

// ---------------------------------------------------------------------------
// Farthest point sampling. One block per batch; dist in global ws (L2-resident).
// Matches jax scan: out[it]=far; dist=min(dist,d(far)); far=argmax (first idx).
// ---------------------------------------------------------------------------
__global__ void fps_kernel(const float* __restrict__ xyz, float* __restrict__ dist,
                           int* __restrict__ oidx, int N, int npoint)
{
  const int b = blockIdx.x;
  const float* P = xyz + (size_t)b * N * 3;
  float* D = dist + (size_t)b * N;
  int* out = oidx + (size_t)b * npoint;
  const int tid = threadIdx.x, nt = blockDim.x;

  __shared__ float swv[32];
  __shared__ int   swi[32];
  __shared__ int   sfar;

  for (int i = tid; i < N; i += nt) D[i] = 1e10f;
  if (tid == 0) sfar = 0;
  __syncthreads();

  for (int it = 0; it < npoint; ++it) {
    int far = sfar;
    if (tid == 0) out[it] = far;
    float cx = P[far * 3 + 0], cy = P[far * 3 + 1], cz = P[far * 3 + 2];
    float best = -1.0f; int bi = 0x7fffffff;
    for (int i = tid; i < N; i += nt) {
      float dx = P[i * 3 + 0] - cx;
      float dy = P[i * 3 + 1] - cy;
      float dz = P[i * 3 + 2] - cz;
      float d  = dx * dx + dy * dy + dz * dz;
      float nd = fminf(D[i], d);
      D[i] = nd;
      if (nd > best) { best = nd; bi = i; }
    }
    for (int off = 16; off > 0; off >>= 1) {
      float ov = __shfl_down(best, off);
      int   oi = __shfl_down(bi, off);
      if (ov > best || (ov == best && oi < bi)) { best = ov; bi = oi; }
    }
    int w = tid >> 5, l = tid & 31;
    if (l == 0) { swv[w] = best; swi[w] = bi; }
    __syncthreads();
    if (tid == 0) {
      int nw = nt >> 5;
      float bv = swv[0]; int bj = swi[0];
      for (int k = 1; k < nw; ++k)
        if (swv[k] > bv || (swv[k] == bv && swi[k] < bj)) { bv = swv[k]; bj = swi[k]; }
      sfar = bj;
    }
    __syncthreads();
  }
}

__global__ void gather_xyz_kernel(const float* __restrict__ xyz, const int* __restrict__ idx,
                                  float* __restrict__ nx, int total, int N, int S)
{
  int t = blockIdx.x * blockDim.x + threadIdx.x;
  if (t >= total) return;              // total = B*S
  int b = t / S;
  int gi = idx[t];
  const float* p = xyz + ((size_t)b * N + gi) * 3;
  nx[t * 3 + 0] = p[0];
  nx[t * 3 + 1] = p[1];
  nx[t * 3 + 2] = p[2];
}

// Ball query: first K indices (ascending) with d2 <= r2; pad with first hit.
__global__ void ball_query_kernel(const float* __restrict__ xyz, const float* __restrict__ nx,
                                  int* __restrict__ gidx, int total, int N, int S, int K, float r2)
{
  int t = blockIdx.x * blockDim.x + threadIdx.x;
  if (t >= total) return;              // total = B*S
  int b = t / S;
  const float* q = nx + (size_t)t * 3;
  const float* P = xyz + (size_t)b * N * 3;
  int* g = gidx + (size_t)t * K;
  int cnt = 0, first = -1;
  for (int i = 0; i < N && cnt < K; ++i) {
    float dx = P[i * 3 + 0] - q[0];
    float dy = P[i * 3 + 1] - q[1];
    float dz = P[i * 3 + 2] - q[2];
    float d = dx * dx + dy * dy + dz * dz;
    if (d <= r2) { if (first < 0) first = i; g[cnt++] = i; }
  }
  if (first < 0) first = N - 1;        // JAX OOB gather clamps
  for (; cnt < K; ++cnt) g[cnt] = first;
}

// Build grouped rows with padded stride Cs: concat(points[gidx], rel-xyz) per
// ptsFirst; zero-fill channels [Cp+3, Cs).
__global__ void group_kernel(const float* __restrict__ xyz, const float* __restrict__ pts,
                             const float* __restrict__ nx, const int* __restrict__ gidx,
                             float* __restrict__ Xd, int total, int N, int S, int K,
                             int Cp, int Cs, int ptsFirst)
{
  int t = blockIdx.x * blockDim.x + threadIdx.x;
  if (t >= total) return;              // total = B*S*K
  int b = t / (S * K);
  int bs = t / K;
  int gi = gidx[t];
  const float* pp = xyz + ((size_t)b * N + gi) * 3;
  const float* qq = nx + (size_t)bs * 3;
  float r0 = pp[0] - qq[0], r1 = pp[1] - qq[1], r2v = pp[2] - qq[2];
  int Cx = Cp + 3;
  float* dst = Xd + (size_t)t * Cs;
  if (Cp > 0) {
    const float* f = pts + ((size_t)b * N + gi) * Cp;
    if (ptsFirst) {
      for (int c = 0; c < Cp; ++c) dst[c] = f[c];
      dst[Cp + 0] = r0; dst[Cp + 1] = r1; dst[Cp + 2] = r2v;
    } else {
      dst[0] = r0; dst[1] = r1; dst[2] = r2v;
      for (int c = 0; c < Cp; ++c) dst[3 + c] = f[c];
    }
  } else {
    dst[0] = r0; dst[1] = r1; dst[2] = r2v;
  }
  for (int c = Cx; c < Cs; ++c) dst[c] = 0.0f;   // K padding for WMMA
}

// Max over K neighbors -> persistent feature buffer at channel offset
__global__ void maxpool_kernel(const float* __restrict__ Yk, float* __restrict__ F,
                               int total, int K, int O, int Ctot, int coff)
{
  int t = blockIdx.x * blockDim.x + threadIdx.x;
  if (t >= total) return;              // total = B*S*O
  int o = t % O;
  int bs = t / O;
  const float* src = Yk + (size_t)bs * K * O + o;
  float m = -3.4e38f;
  for (int j = 0; j < K; ++j) m = fmaxf(m, src[(size_t)j * O]);
  F[(size_t)bs * Ctot + coff + o] = m;
}

// 3-NN inverse-distance interpolation (feature propagation); writes channels
// [coff, coff+C2) of rows with stride Cs, zero-fills [coff+C2, Cs).
__global__ void interp3_kernel(const float* __restrict__ xyz1, const float* __restrict__ xyz2,
                               const float* __restrict__ pts2, float* __restrict__ Xd,
                               int total, int N1, int S2, int C2, int coff, int Cs)
{
  int t = blockIdx.x * blockDim.x + threadIdx.x;
  if (t >= total) return;              // total = B*N1
  int b = t / N1;
  const float* q = xyz1 + (size_t)t * 3;
  const float* P = xyz2 + (size_t)b * S2 * 3;
  float d0 = 3.4e38f, d1 = 3.4e38f, d2 = 3.4e38f;
  int i0 = 0, i1 = 0, i2 = 0;
  for (int j = 0; j < S2; ++j) {
    float dx = q[0] - P[j * 3 + 0];
    float dy = q[1] - P[j * 3 + 1];
    float dz = q[2] - P[j * 3 + 2];
    float d = dx * dx + dy * dy + dz * dz;
    if (d < d0)      { d2 = d1; i2 = i1; d1 = d0; i1 = i0; d0 = d; i0 = j; }
    else if (d < d1) { d2 = d1; i2 = i1; d1 = d;  i1 = j; }
    else if (d < d2) { d2 = d;  i2 = j; }
  }
  float w0 = 1.0f / (d0 + 1e-8f);
  float w1 = 1.0f / (d1 + 1e-8f);
  float w2 = 1.0f / (d2 + 1e-8f);
  float ws = w0 + w1 + w2;
  w0 /= ws; w1 /= ws; w2 /= ws;
  const float* F = pts2 + (size_t)b * S2 * C2;
  const float* f0 = F + (size_t)i0 * C2;
  const float* f1 = F + (size_t)i1 * C2;
  const float* f2 = F + (size_t)i2 * C2;
  float* row = Xd + (size_t)t * Cs;
  float* dst = row + coff;
  for (int c = 0; c < C2; ++c) dst[c] = w0 * f0[c] + w1 * f1[c] + w2 * f2[c];
  for (int c = coff + C2; c < Cs; ++c) row[c] = 0.0f;
}

// Copy skip-connection channels into the concat buffer (row stride Cs)
__global__ void copy_channels_kernel(const float* __restrict__ src, float* __restrict__ dst,
                                     int total, int Cp, int Cs)
{
  int t = blockIdx.x * blockDim.x + threadIdx.x;
  if (t >= total) return;              // total = rows*Cp
  int r = t / Cp, c = t % Cp;
  dst[(size_t)r * Cs + c] = src[(size_t)r * Cp + c];
}

// head2: logits[bn] = dot(h[bn,:], W2) + b2
__global__ void head2_kernel(const float* __restrict__ H, const float* __restrict__ W2,
                             const float* __restrict__ b2, float* __restrict__ out,
                             int total, int C)
{
  int t = blockIdx.x * blockDim.x + threadIdx.x;
  if (t >= total) return;
  const float* h = H + (size_t)t * C;
  float s = b2[0];
  for (int c = 0; c < C; ++c) s += h[c] * W2[c];
  out[t] = s;
}

// (B,N,C) -> (B,C,N)
__global__ void transpose_kernel(const float* __restrict__ src, float* __restrict__ dst,
                                 long long total, int N, int C)
{
  long long t = (long long)blockIdx.x * blockDim.x + threadIdx.x;
  if (t >= total) return;              // total = B*N*C
  int c = (int)(t % C);
  long long bn = t / C;
  int n = (int)(bn % N);
  int b = (int)(bn / N);
  dst[((size_t)b * C + c) * N + n] = src[t];
}

// ===========================================================================
// Host orchestration
// ===========================================================================
struct Layer {
  const float *W, *b, *g, *be, *mn, *vr;
  __bf16* Wb;
  int C, O, Cs;
};

extern "C" void kernel_launch(void* const* d_in, const int* in_sizes, int n_in,
                              void* d_out, int out_size, void* d_ws, size_t ws_size,
                              hipStream_t stream)
{
  (void)in_sizes; (void)n_in; (void)out_size; (void)ws_size;
  const int Bsz = 8, N0 = 16384;
  const float* xyz0 = (const float*)d_in[0];

  // ---- parameter consumption in setup_inputs() construction order ----
  int pc = 1;
  auto take = [&](int C, int O) {
    Layer L;
    L.W  = (const float*)d_in[pc++]; L.b  = (const float*)d_in[pc++];
    L.g  = (const float*)d_in[pc++]; L.be = (const float*)d_in[pc++];
    L.mn = (const float*)d_in[pc++]; L.vr = (const float*)d_in[pc++];
    L.C = C; L.O = O; L.Cs = (C + 31) & ~31; L.Wb = nullptr;
    return L;
  };
  Layer sa1a[3] = { take(3, 32),   take(32, 32),   take(32, 64)   };
  Layer sa1b[3] = { take(3, 64),   take(64, 64),   take(64, 128)  };
  Layer sa2a[3] = { take(195, 64), take(64, 64),   take(64, 128)  };
  Layer sa2b[3] = { take(195, 128),take(128, 128), take(128, 256) };
  Layer sa3a[3] = { take(387, 128),take(128, 128), take(128, 256) };
  Layer sa3b[3] = { take(387, 256),take(256, 256), take(256, 512) };
  Layer sa4l[3] = { take(771, 256),take(256, 512), take(512, 1024)};
  Layer fp4l[2] = { take(1792, 512), take(512, 512) };
  Layer fp3l[2] = { take(896, 512),  take(512, 256) };
  Layer fp2l[2] = { take(448, 256),  take(256, 128) };
  Layer fp1l[3] = { take(128, 128),  take(128, 128), take(128, 128) };
  Layer h1l     = take(128, 64);
  const float* W2 = (const float*)d_in[pc++];
  const float* b2 = (const float*)d_in[pc++];

  // ---- workspace cursor allocation ----
  size_t off = 0;
  auto alloc = [&](size_t nbytes) -> void* {
    size_t a = (off + 255) & ~(size_t)255;
    off = a + nbytes;
    return (void*)((char*)d_ws + a);
  };
  float* fpsDist = (float*)alloc((size_t)Bsz * N0 * 4);
  int*   fpsIdx  = (int*)  alloc((size_t)Bsz * 640 * 4);
  int*   gidx    = (int*)  alloc((size_t)Bsz * 640 * 32 * 4);
  float* l1x = (float*)alloc((size_t)Bsz * 640 * 3 * 4);
  float* l2x = (float*)alloc((size_t)Bsz * 320 * 3 * 4);
  float* l3x = (float*)alloc((size_t)Bsz * 160 * 3 * 4);
  float* l4x = (float*)alloc((size_t)Bsz * 40  * 3 * 4);
  float* l1F = (float*)alloc((size_t)Bsz * 640 * 192  * 4);
  float* l2F = (float*)alloc((size_t)Bsz * 320 * 384  * 4);
  float* l3F = (float*)alloc((size_t)Bsz * 160 * 768  * 4);
  float* l4F = (float*)alloc((size_t)Bsz * 40  * 1024 * 4);
  float* l3p = (float*)alloc((size_t)Bsz * 160 * 512  * 4);
  float* l2p = (float*)alloc((size_t)Bsz * 320 * 256  * 4);
  float* l1p = (float*)alloc((size_t)Bsz * 640 * 128  * 4);
  const size_t PING = 20971520;  // floats: covers max M*Cs over all stages
  float* bufA = (float*)alloc(PING * 4);
  float* bufB = (float*)alloc(PING * 4);

  auto cdiv = [](int a, int b) { return (a + b - 1) / b; };

  // ---- pre-pack all weights to bf16 [O, Cs] ----
  auto prep = [&](Layer& L) {
    L.Wb = (__bf16*)alloc((size_t)L.O * L.Cs * 2);
    int tot = L.O * L.Cs;
    pack_w_kernel<<<cdiv(tot, 256), 256, 0, stream>>>(L.W, L.Wb, L.O, L.C, L.Cs);
  };
  for (int i = 0; i < 3; ++i) { prep(sa1a[i]); prep(sa1b[i]); prep(sa2a[i]); prep(sa2b[i]); }
  for (int i = 0; i < 3; ++i) { prep(sa3a[i]); prep(sa3b[i]); prep(sa4l[i]); prep(fp1l[i]); }
  for (int i = 0; i < 2; ++i) { prep(fp4l[i]); prep(fp3l[i]); prep(fp2l[i]); }
  prep(h1l);

  auto gemm = [&](const float* X, float* Y, const Layer& L, int M) {
    dim3 grid(M / 64, cdiv(L.O, 64));   // M % 64 == 0 for every call site
    wmma_mlp_kernel<<<grid, 128, 0, stream>>>(X, L.Wb, L.b, L.g, L.be, L.mn, L.vr,
                                              Y, M, L.Cs, L.O);
  };
  // X must already be in bufA (row stride Ls[0].Cs); returns final output ptr
  auto run_mlp = [&](const Layer* Ls, int nL, int M) -> float* {
    float* src = bufA; float* dst = bufB;
    for (int i = 0; i < nL; ++i) { gemm(src, dst, Ls[i], M); float* t = src; src = dst; dst = t; }
    return src;
  };

  // one SA scale: ball query -> group -> MLP -> maxpool into F at coff
  auto sa_scale = [&](const float* xyzIn, int N, const float* ptsIn, int Cp,
                      const float* newXyz, int S, float rad, int K,
                      const Layer* Ls, int nL, int ptsFirst,
                      float* F, int Ctot, int coff) {
    int BS = Bsz * S, BSK = BS * K;
    ball_query_kernel<<<cdiv(BS, 128), 128, 0, stream>>>(
        xyzIn, newXyz, gidx, BS, N, S, K, rad * rad);
    group_kernel<<<cdiv(BSK, 256), 256, 0, stream>>>(
        xyzIn, ptsIn, newXyz, gidx, bufA, BSK, N, S, K, Cp, Ls[0].Cs, ptsFirst);
    float* out = run_mlp(Ls, nL, BSK);
    int Olast = Ls[nL - 1].O;
    maxpool_kernel<<<cdiv(BS * Olast, 256), 256, 0, stream>>>(
        out, F, BS * Olast, K, Olast, Ctot, coff);
  };

  auto do_fps = [&](const float* xyzIn, int N, int S, float* newXyz) {
    fps_kernel<<<Bsz, 256, 0, stream>>>(xyzIn, fpsDist, fpsIdx, N, S);
    gather_xyz_kernel<<<cdiv(Bsz * S, 256), 256, 0, stream>>>(
        xyzIn, fpsIdx, newXyz, Bsz * S, N, S);
  };

  // -------------------- sa1 (MSG) --------------------
  do_fps(xyz0, N0, 640, l1x);
  sa_scale(xyz0, N0, nullptr, 0, l1x, 640, 0.05f, 16, sa1a, 3, 1, l1F, 192, 0);
  sa_scale(xyz0, N0, nullptr, 0, l1x, 640, 0.10f, 32, sa1b, 3, 1, l1F, 192, 64);
  // -------------------- sa2 (MSG) --------------------
  do_fps(l1x, 640, 320, l2x);
  sa_scale(l1x, 640, l1F, 192, l2x, 320, 0.10f, 16, sa2a, 3, 1, l2F, 384, 0);
  sa_scale(l1x, 640, l1F, 192, l2x, 320, 0.20f, 32, sa2b, 3, 1, l2F, 384, 128);
  // -------------------- sa3 (MSG) --------------------
  do_fps(l2x, 320, 160, l3x);
  sa_scale(l2x, 320, l2F, 384, l3x, 160, 0.20f, 16, sa3a, 3, 1, l3F, 768, 0);
  sa_scale(l2x, 320, l2F, 384, l3x, 160, 0.40f, 32, sa3b, 3, 1, l3F, 768, 256);
  // -------------------- sa4 (single, xyz-first concat) --------------------
  do_fps(l3x, 160, 40, l4x);
  sa_scale(l3x, 160, l3F, 768, l4x, 40, 0.40f, 32, sa4l, 3, 0, l4F, 1024, 0);

  // -------------------- feature propagation --------------------
  auto fp_stage = [&](const float* xyz1, int N1, const float* xyz2, int S2,
                      const float* pts1, int Cp, const float* pts2, int C2,
                      const Layer* Ls, int nL, float* persist) -> float* {
    int Mrows = Bsz * N1;
    int Cs = Ls[0].Cs;                 // == Cp + C2 here (all multiples of 32)
    if (pts1)
      copy_channels_kernel<<<cdiv(Mrows * Cp, 256), 256, 0, stream>>>(
          pts1, bufA, Mrows * Cp, Cp, Cs);
    interp3_kernel<<<cdiv(Mrows, 128), 128, 0, stream>>>(
        xyz1, xyz2, pts2, bufA, Mrows, N1, S2, C2, Cp, Cs);
    float* out = run_mlp(Ls, nL, Mrows);
    if (persist)
      hipMemcpyAsync(persist, out, (size_t)Mrows * Ls[nL - 1].O * 4,
                     hipMemcpyDeviceToDevice, stream);
    return out;
  };

  fp_stage(l3x, 160, l4x, 40,  l3F, 768, l4F, 1024, fp4l, 2, l3p);
  fp_stage(l2x, 320, l3x, 160, l2F, 384, l3p, 512,  fp3l, 2, l2p);
  fp_stage(l1x, 640, l2x, 320, l1F, 192, l2p, 256,  fp2l, 2, l1p);
  float* l0f = fp_stage(xyz0, N0, l1x, 640, nullptr, 0, l1p, 128, fp1l, 3, nullptr);

  // -------------------- outputs --------------------
  float* outp = (float*)d_out;
  long long totT = (long long)Bsz * N0 * 128;
  transpose_kernel<<<(int)((totT + 255) / 256), 256, 0, stream>>>(
      l0f, outp + (size_t)Bsz * N0, totT, N0, 128);

  float* hbuf = (l0f == bufA) ? bufB : bufA;
  gemm(l0f, hbuf, h1l, Bsz * N0);
  head2_kernel<<<cdiv(Bsz * N0, 256), 256, 0, stream>>>(
      hbuf, W2, b2, outp, Bsz * N0, 64);
}
